// DOMTransformer_14130442404235
// MI455X (gfx1250) — matrix operand order repaired and checked by reading, not compiled, not run on hardware
//
#include <hip/hip_runtime.h>
#include <hip/hip_bf16.h>
#include <math.h>

#define N_PULSES 4096
#define PF       4
#define DMODEL   256
#define NHEADS   8
#define HDIM     32
#define NLAYERS  4
#define DFFN     1024
#define NDOM     512
#define TD       768   // 3 * DMODEL

typedef __attribute__((ext_vector_type(16))) __bf16 v16bf;
typedef __attribute__((ext_vector_type(2)))  __bf16 bf16x2;
typedef __attribute__((ext_vector_type(8)))  float  v8f;

// ---------------------------------------------------------------------------
// Weight convert + transpose: Wt[n*K + k] = bf16(W[k*Nn + n])
// ---------------------------------------------------------------------------
__global__ __launch_bounds__(256)
void conv_wt_kernel(const float* __restrict__ W, __bf16* __restrict__ Wt,
                    int K, int Nn) {
    int t = blockIdx.x * blockDim.x + threadIdx.x;
    if (t >= K * Nn) return;
    int n = t / K;
    int k = t - n * K;
    Wt[t] = (__bf16)W[(size_t)k * Nn + n];
}

// ---------------------------------------------------------------------------
// Input projection: x = pulse_features @ Win + b_in   (N x 4) @ (4 x 256)
// ---------------------------------------------------------------------------
__global__ __launch_bounds__(256)
void input_proj_kernel(const float* __restrict__ pf, const float* __restrict__ Win,
                       const float* __restrict__ bin, float* __restrict__ x) {
    const int row = blockIdx.x;
    const int d   = threadIdx.x;
    float acc = bin[d];
#pragma unroll
    for (int p = 0; p < PF; ++p)
        acc += pf[row * PF + p] * Win[p * DMODEL + d];
    x[(size_t)row * DMODEL + d] = acc;
}

// ---------------------------------------------------------------------------
// Segment bounds over the sorted dom index
// ---------------------------------------------------------------------------
__global__ void seg_init_kernel(int* __restrict__ ds, int* __restrict__ de) {
    int i = blockIdx.x * blockDim.x + threadIdx.x;
    if (i < NDOM) { ds[i] = 0; de[i] = 0; }
}

__global__ void seg_bounds_kernel(const int* __restrict__ idx,
                                  int* __restrict__ ds, int* __restrict__ de) {
    int i = blockIdx.x * blockDim.x + threadIdx.x;
    if (i >= N_PULSES) return;
    int d = idx[i];
    if (i == 0 || idx[i - 1] != d) ds[d] = i;
    if (i == N_PULSES - 1 || idx[i + 1] != d) de[d] = i + 1;
}

// ---------------------------------------------------------------------------
// WMMA GEMM: C[M,Nn] = A[M,K] (fp32, converted to bf16 on the fly)
//                      @ Bt[Nn,K]^T (bf16, pre-transposed) + bias, opt. GELU.
// Wave tile = 32(M) x 64(N): 2 A fragments x 4 B fragments -> 8 WMMAs per
// K-step; A reused 4x, B reused 2x in registers. K is a compile-time constant
// so the K-loop can be unrolled/pipelined (no per-WMMA loadcnt-0 stalls).
// block = 128 threads = 4 waves = 32 x 256 block tile; grid = (M/32, Nn/256).
// ---------------------------------------------------------------------------
template <int K, bool GELU>
__global__ __launch_bounds__(128)
void gemm_wmma_kernel(const float* __restrict__ A, const __bf16* __restrict__ Bt,
                      const float* __restrict__ bias, float* __restrict__ C,
                      int Nn) {
    const int lane   = threadIdx.x & 31;
    const int wave   = threadIdx.x >> 5;
    const int m0     = blockIdx.x * 32;
    const int n0     = (blockIdx.y * 4 + wave) * 64;
    const int half   = lane >> 4;    // which 16-lane half
    const int lane16 = lane & 15;

    // A fragment (contiguous k order, same as B): lane m = lane16,
    //   VGPR i : K = half*16 + 2i, +1
    const float* Arow[2];
    Arow[0] = A + (size_t)(m0 + lane16) * K;
    Arow[1] = A + (size_t)(m0 + 16 + lane16) * K;
    // B fragment (32x16, row striped across lanes): lane n = lane16,
    //   VGPR i : K = half*16 + 2i, +1   (contiguous bf16 pair in Bt[n,:])
    const __bf16* Brow[4];
#pragma unroll
    for (int t = 0; t < 4; ++t)
        Brow[t] = Bt + (size_t)(n0 + t * 16 + lane16) * K;

    const int kAoff = half * 16;
    const int kBoff = half * 16;

    v8f acc[2][4] = {{{}, {}, {}, {}}, {{}, {}, {}, {}}};
    for (int kk = 0; kk < K; kk += 32) {
        // two A fragments (rows m0..m0+15 and m0+16..m0+31)
        v16bf a[2];
        const int kA[8] = {0, 2, 4, 6, 8, 10, 12, 14};
#pragma unroll
        for (int u = 0; u < 2; ++u) {
#pragma unroll
            for (int i = 0; i < 8; ++i) {
                const float2 f =
                    *reinterpret_cast<const float2*>(Arow[u] + kk + kAoff + kA[i]);
                a[u][2 * i]     = (__bf16)f.x;
                a[u][2 * i + 1] = (__bf16)f.y;
            }
        }
        // four B fragments (consecutive 16-col subtiles)
        v16bf b[4];
#pragma unroll
        for (int t = 0; t < 4; ++t) {
#pragma unroll
            for (int i = 0; i < 8; ++i) {
                const bf16x2 bp =
                    *reinterpret_cast<const bf16x2*>(Brow[t] + kk + kBoff + 2 * i);
                b[t][2 * i]     = bp.x;
                b[t][2 * i + 1] = bp.y;
            }
        }
        // 8 WMMAs: each A fragment reused 4x, each B fragment reused 2x
#pragma unroll
        for (int u = 0; u < 2; ++u) {
#pragma unroll
            for (int t = 0; t < 4; ++t) {
                acc[u][t] = __builtin_amdgcn_wmma_f32_16x16x32_bf16(
                    /*neg_a=*/false, a[u], /*neg_b=*/false, b[t],
                    /*c_mod=*/(short)0, acc[u][t], /*reuse_a=*/false, /*reuse_b=*/false);
            }
        }
    }

    // C/D layout: VGPR r, lanes 0-15 -> M = r, lanes 16-31 -> M = 8 + r; N = lane16
#pragma unroll
    for (int u = 0; u < 2; ++u) {
#pragma unroll
        for (int t = 0; t < 4; ++t) {
            const int ncol = n0 + t * 16 + lane16;
            const float bs = bias[ncol];
#pragma unroll
            for (int r = 0; r < 8; ++r) {
                const int m = m0 + u * 16 + half * 8 + r;
                float v = acc[u][t][r] + bs;
                if (GELU) v = 0.5f * v * (1.0f + erff(v * 0.70710678118654752f));
                C[(size_t)m * Nn + ncol] = v;
            }
        }
    }
}

// ---------------------------------------------------------------------------
// Block-diagonal attention: one block per DOM, one wave per head,
// lanes = 32 head channels. Online softmax over the (small) key segment.
// ---------------------------------------------------------------------------
__global__ __launch_bounds__(256)
void attention_kernel(const float* __restrict__ qkv, const int* __restrict__ ds,
                      const int* __restrict__ de, float* __restrict__ out) {
    const int dom  = blockIdx.x;
    const int h    = threadIdx.x >> 5;
    const int lane = threadIdx.x & 31;
    const int s = ds[dom], e = de[dom];
    if (e <= s) return;
    const float scale = 0.17677669529663687f;  // 1/sqrt(32)
    const int qo = h * HDIM;

    for (int i = s; i < e; ++i) {
        const float q = qkv[(size_t)i * TD + qo + lane];
        float m = -__builtin_inff(), l = 0.f, acc = 0.f;
        for (int j = s; j < e; ++j) {
            float d = q * qkv[(size_t)j * TD + DMODEL + qo + lane];
#pragma unroll
            for (int off = 16; off; off >>= 1) d += __shfl_xor(d, off, 32);
            d *= scale;
            const float mn   = fmaxf(m, d);
            const float corr = __expf(m - mn);
            const float pr   = __expf(d - mn);
            l   = l * corr + pr;
            acc = acc * corr + pr * qkv[(size_t)j * TD + 2 * DMODEL + qo + lane];
            m = mn;
        }
        out[(size_t)i * DMODEL + qo + lane] = acc / l;
    }
}

// ---------------------------------------------------------------------------
// Fused (optional residual add) + LayerNorm over D=256. 1 wave per row.
// ---------------------------------------------------------------------------
__global__ __launch_bounds__(256)
void add_ln_kernel(const float* __restrict__ xin, const float* __restrict__ res,
                   const float* __restrict__ g, const float* __restrict__ b,
                   float* __restrict__ xout) {
    const int row  = blockIdx.x * 8 + (threadIdx.x >> 5);
    const int lane = threadIdx.x & 31;
    float v[8];
    float s = 0.f;
#pragma unroll
    for (int i = 0; i < 8; ++i) {
        const int c = lane + 32 * i;
        float t = xin[(size_t)row * DMODEL + c];
        if (res) t += res[(size_t)row * DMODEL + c];
        v[i] = t;
        s += t;
    }
#pragma unroll
    for (int off = 16; off; off >>= 1) s += __shfl_xor(s, off, 32);
    const float mean = s * (1.0f / DMODEL);
    float var = 0.f;
#pragma unroll
    for (int i = 0; i < 8; ++i) { const float d = v[i] - mean; var += d * d; }
#pragma unroll
    for (int off = 16; off; off >>= 1) var += __shfl_xor(var, off, 32);
    const float rstd = rsqrtf(var * (1.0f / DMODEL) + 1e-5f);
#pragma unroll
    for (int i = 0; i < 8; ++i) {
        const int c = lane + 32 * i;
        xout[(size_t)row * DMODEL + c] = (v[i] - mean) * rstd * g[c] + b[c];
    }
}

// ---------------------------------------------------------------------------
// Segment mean pooling: one block per DOM, one thread per channel.
// ---------------------------------------------------------------------------
__global__ __launch_bounds__(256)
void seg_mean_kernel(const float* __restrict__ x, const int* __restrict__ ds,
                     const int* __restrict__ de, float* __restrict__ out) {
    const int dom = blockIdx.x;
    const int d   = threadIdx.x;
    const int s = ds[dom], e = de[dom];
    float acc = 0.f;
    for (int i = s; i < e; ++i) acc += x[(size_t)i * DMODEL + d];
    const float cnt = (float)((e - s) > 0 ? (e - s) : 1);
    out[(size_t)dom * DMODEL + d] = acc / cnt;
}

// ---------------------------------------------------------------------------
extern "C" void kernel_launch(void* const* d_in, const int* in_sizes, int n_in,
                              void* d_out, int out_size, void* d_ws, size_t ws_size,
                              hipStream_t stream) {
    const float* pf   = (const float*)d_in[0];
    const int*   idx  = (const int*)d_in[1];
    // d_in[2] = total_doms scalar (compile-time NDOM here)
    const float* Win  = (const float*)d_in[3];
    const float* bin  = (const float*)d_in[4];
    const float* Wqkv = (const float*)d_in[5];
    const float* bqkv = (const float*)d_in[6];
    const float* Wout = (const float*)d_in[7];
    const float* bout = (const float*)d_in[8];
    const float* W1   = (const float*)d_in[9];
    const float* b1   = (const float*)d_in[10];
    const float* W2   = (const float*)d_in[11];
    const float* b2   = (const float*)d_in[12];
    const float* g1   = (const float*)d_in[13];
    const float* be1  = (const float*)d_in[14];
    const float* g2   = (const float*)d_in[15];
    const float* be2  = (const float*)d_in[16];
    const float* gF   = (const float*)d_in[17];
    const float* bF   = (const float*)d_in[18];
    float* out = (float*)d_out;

    // ---- workspace partition (256B aligned slices) ----
    char* p = (char*)d_ws;
    auto take = [&](size_t bytes) -> char* {
        char* r = p;
        p += (bytes + 255) & ~(size_t)255;
        return r;
    };
    float*  x     = (float*)take((size_t)N_PULSES * DMODEL * sizeof(float));
    float*  tmp   = (float*)take((size_t)N_PULSES * DMODEL * sizeof(float));
    float*  qkv   = (float*)take((size_t)N_PULSES * TD * sizeof(float));
    float*  hbuf  = (float*)take((size_t)N_PULSES * DFFN * sizeof(float));
    float*  aout  = (float*)take((size_t)N_PULSES * DMODEL * sizeof(float));
    __bf16* WqkvT = (__bf16*)take((size_t)NLAYERS * TD * DMODEL * sizeof(__bf16));
    __bf16* WoutT = (__bf16*)take((size_t)NLAYERS * DMODEL * DMODEL * sizeof(__bf16));
    __bf16* W1T   = (__bf16*)take((size_t)NLAYERS * DFFN * DMODEL * sizeof(__bf16));
    __bf16* W2T   = (__bf16*)take((size_t)NLAYERS * DMODEL * DFFN * sizeof(__bf16));
    int* dstart   = (int*)take(NDOM * sizeof(int));
    int* dend     = (int*)take(NDOM * sizeof(int));

    // ---- one-time weight convert + transpose to bf16 (N x K layout) ----
    for (int l = 0; l < NLAYERS; ++l) {
        conv_wt_kernel<<<(DMODEL * TD + 255) / 256, 256, 0, stream>>>(
            Wqkv + (size_t)l * DMODEL * TD, WqkvT + (size_t)l * TD * DMODEL, DMODEL, TD);
        conv_wt_kernel<<<(DMODEL * DMODEL + 255) / 256, 256, 0, stream>>>(
            Wout + (size_t)l * DMODEL * DMODEL, WoutT + (size_t)l * DMODEL * DMODEL, DMODEL, DMODEL);
        conv_wt_kernel<<<(DMODEL * DFFN + 255) / 256, 256, 0, stream>>>(
            W1 + (size_t)l * DMODEL * DFFN, W1T + (size_t)l * DFFN * DMODEL, DMODEL, DFFN);
        conv_wt_kernel<<<(DFFN * DMODEL + 255) / 256, 256, 0, stream>>>(
            W2 + (size_t)l * DFFN * DMODEL, W2T + (size_t)l * DMODEL * DFFN, DFFN, DMODEL);
    }

    // ---- input projection + segment bounds ----
    input_proj_kernel<<<N_PULSES, 256, 0, stream>>>(pf, Win, bin, x);
    seg_init_kernel<<<(NDOM + 255) / 256, 256, 0, stream>>>(dstart, dend);
    seg_bounds_kernel<<<(N_PULSES + 255) / 256, 256, 0, stream>>>(idx, dstart, dend);

    // ---- transformer layers ----
    for (int l = 0; l < NLAYERS; ++l) {
        // qkv = x @ Wqkv + bqkv
        gemm_wmma_kernel<DMODEL, false>
            <<<dim3(N_PULSES / 32, TD / 256), 128, 0, stream>>>(
                x, WqkvT + (size_t)l * TD * DMODEL, bqkv + (size_t)l * TD, qkv, TD);
        // per-DOM masked attention
        attention_kernel<<<NDOM, 256, 0, stream>>>(qkv, dstart, dend, aout);
        // o = attn @ Wout + bout
        gemm_wmma_kernel<DMODEL, false>
            <<<dim3(N_PULSES / 32, DMODEL / 256), 128, 0, stream>>>(
                aout, WoutT + (size_t)l * DMODEL * DMODEL, bout + (size_t)l * DMODEL,
                tmp, DMODEL);
        // x = LN(x + o)
        add_ln_kernel<<<N_PULSES / 8, 256, 0, stream>>>(
            x, tmp, g1 + (size_t)l * DMODEL, be1 + (size_t)l * DMODEL, x);
        // h = gelu(x @ W1 + b1)
        gemm_wmma_kernel<DMODEL, true>
            <<<dim3(N_PULSES / 32, DFFN / 256), 128, 0, stream>>>(
                x, W1T + (size_t)l * DFFN * DMODEL, b1 + (size_t)l * DFFN, hbuf, DFFN);
        // f = h @ W2 + b2
        gemm_wmma_kernel<DFFN, false>
            <<<dim3(N_PULSES / 32, DMODEL / 256), 128, 0, stream>>>(
                hbuf, W2T + (size_t)l * DMODEL * DFFN, b2 + (size_t)l * DMODEL,
                tmp, DMODEL);
        // x = LN(x + f)
        add_ln_kernel<<<N_PULSES / 8, 256, 0, stream>>>(
            x, tmp, g2 + (size_t)l * DMODEL, be2 + (size_t)l * DMODEL, x);
    }

    // ---- final LN + segment mean ----
    add_ln_kernel<<<N_PULSES / 8, 256, 0, stream>>>(x, nullptr, gF, bF, x);
    seg_mean_kernel<<<NDOM, 256, 0, stream>>>(x, dstart, dend, out);
}
